// DeformableTransformerDecoderLayer_59279138619681
// MI455X (gfx1250) — matrix operand order repaired and checked by reading, not compiled, not run on hardware
//
#include <hip/hip_runtime.h>

// ---------------- problem constants ----------------
#define B2v   10      // N*NF
#define LQv   300
#define Cv    256
#define NHv   8
#define HDv   32
#define Sv    11253
#define DFFv  1024
#define Mq    3000    // B2*LQ
#define Mval  112530  // B2*S

typedef __attribute__((ext_vector_type(16))) __bf16          v16bf;
typedef __attribute__((ext_vector_type(8)))  float           v8f;
typedef __attribute__((ext_vector_type(16))) unsigned short  v16u;
typedef __attribute__((ext_vector_type(8)))  unsigned short  v8u;

__device__ __forceinline__ unsigned short f2bf(float f) {
  unsigned u = __float_as_uint(f);
  return (unsigned short)((u + 0x7FFFu + ((u >> 16) & 1u)) >> 16); // RNE truncate
}

// ---------------------------------------------------------------------------
// GEMM: out[M,N] = A[M,K] @ W[N,K]^T + bias[N], optional ReLU.
// fp32 in/out, bf16 WMMA, register-held double-buffered LDS pipeline:
//   iter k: [issue global loads tile k+1] -> [WMMA from LDS buf k] ->
//           [wait loads, cvt, ds_store buf k^1] -> barrier
// (ds ops are in-order per wave, so stores MUST follow the frag loads.)
//   <64,256,*>: BM=64, BN=256 — waves 2x4, per-wave 32x64 (2x4 WMMA tiles);
//               whole N in one block column -> A streams from HBM exactly once.
//   <128,64,*>: BM=128, BN=64 — waves 4x2, per-wave 32x32 (2x2 WMMA tiles).
// ---------------------------------------------------------------------------
template<int BM, int BN, int RELU>
__global__ __launch_bounds__(256) void gemm_bf16_kernel(
    const float* __restrict__ A, const float* __restrict__ W,
    const float* __restrict__ bias, float* __restrict__ out,
    int M, int N, int K) {
  constexpr int WROWS = BM / 32;          // waves along M
  constexpr int WCOLS = 8 / WROWS;        // waves along N
  constexpr int TI = 2;                   // 16-row tiles per wave (32 rows)
  constexpr int TJ = BN / (WCOLS * 16);   // 16-col tiles per wave
  constexpr int AIT = BM / 32;            // A-tile load iters per thread
  constexpr int BIT = BN / 32;            // B-tile load iters per thread
  static_assert(WROWS * 32 == BM && WCOLS * TJ * 16 == BN, "tile config");

  __shared__ unsigned short As[2][BM][40];  // [buf][m][k], pitch 40: 16B chunks stay aligned
  __shared__ unsigned short Bs[2][BN][40];  // [buf][n][k]

  const int tid  = threadIdx.x;
  const int m0   = blockIdx.y * BM;
  const int n0   = blockIdx.x * BN;
  const int w    = tid >> 5, lane = tid & 31;
  const int wr   = w % WROWS, wc = w / WROWS;
  const int half = lane >> 4, lm = lane & 15;

  v8f acc[TI][TJ];
  #pragma unroll
  for (int i = 0; i < TI; i++)
    #pragma unroll
    for (int j = 0; j < TJ; j++) {
      float bv = bias[n0 + wc * (TJ * 16) + j * 16 + lm];
      #pragma unroll
      for (int e = 0; e < 8; e++) acc[i][j][e] = bv;
    }

  const int lr = tid >> 3;         // 0..31 (row within a 32-row slice)
  const int lc = (tid & 7) * 4;    // 0,4,...,28 (k offset, float4)

  float4 fa[AIT], fb[BIT];

  auto load_tiles = [&](int k0) {
    #pragma unroll
    for (int it = 0; it < AIT; it++) {
      int gm = m0 + lr + it * 32; if (gm > M - 1) gm = M - 1;
      const float* ap = A + (size_t)gm * K + k0 + lc;
      fa[it] = *reinterpret_cast<const float4*>(ap);
      __builtin_prefetch(ap + 64, 0, 1);     // speculative, distance 2 K-tiles
    }
    #pragma unroll
    for (int it = 0; it < BIT; it++) {
      int gn = n0 + lr + it * 32; if (gn > N - 1) gn = N - 1;
      fb[it] = *reinterpret_cast<const float4*>(W + (size_t)gn * K + k0 + lc);
    }
  };
  auto store_tiles = [&](int buf) {
    #pragma unroll
    for (int it = 0; it < AIT; it++) {
      int r = lr + it * 32;
      As[buf][r][lc + 0] = f2bf(fa[it].x); As[buf][r][lc + 1] = f2bf(fa[it].y);
      As[buf][r][lc + 2] = f2bf(fa[it].z); As[buf][r][lc + 3] = f2bf(fa[it].w);
    }
    #pragma unroll
    for (int it = 0; it < BIT; it++) {
      int r = lr + it * 32;
      Bs[buf][r][lc + 0] = f2bf(fb[it].x); Bs[buf][r][lc + 1] = f2bf(fb[it].y);
      Bs[buf][r][lc + 2] = f2bf(fb[it].z); Bs[buf][r][lc + 3] = f2bf(fb[it].w);
    }
  };

  // prologue: stage tile 0
  load_tiles(0);
  store_tiles(0);
  __syncthreads();

  const int nkt = K >> 5;
  for (int kt = 0; kt < nkt; kt++) {
    const int buf = kt & 1;
    if (kt + 1 < nkt) load_tiles((kt + 1) << 5);   // loadcnt ticks during WMMA

    v16bf af[TI], bfr[TJ];
    #pragma unroll
    for (int i = 0; i < TI; i++) {            // ISA 16-bit A 16x32 layout
      int row = wr * 32 + i * 16 + lm;
      v8u c0 = *reinterpret_cast<const v8u*>(&As[buf][row][8 * half]);
      v8u c1 = *reinterpret_cast<const v8u*>(&As[buf][row][16 + 8 * half]);
      v16u t;
      #pragma unroll
      for (int e = 0; e < 8; e++) { t[e] = c0[e]; t[e + 8] = c1[e]; }
      af[i] = __builtin_bit_cast(v16bf, t);
    }
    #pragma unroll
    for (int j = 0; j < TJ; j++) {            // B: lane = n, half selects K group
      int nn = wc * (TJ * 16) + j * 16 + lm;
      v8u c0 = *reinterpret_cast<const v8u*>(&Bs[buf][nn][8 * half]);
      v8u c1 = *reinterpret_cast<const v8u*>(&Bs[buf][nn][16 + 8 * half]);
      v16u t;
      #pragma unroll
      for (int e = 0; e < 8; e++) { t[e] = c0[e]; t[e + 8] = c1[e]; }
      bfr[j] = __builtin_bit_cast(v16bf, t);
    }
    #pragma unroll
    for (int i = 0; i < TI; i++)
      #pragma unroll
      for (int j = 0; j < TJ; j++)
        acc[i][j] = __builtin_amdgcn_wmma_f32_16x16x32_bf16(
            false, af[i], false, bfr[j], (short)0, acc[i][j], false, false);

    if (kt + 1 < nkt) store_tiles(buf ^ 1);   // waits loads here, after compute
    __syncthreads();
  }

  // ---- epilogue ----
  const int ncol = n0 + wc * (TJ * 16) + lm;
  if (m0 + BM <= M) {
    // fast path: full tile, unguarded, incremental addressing
    #pragma unroll
    for (int i = 0; i < TI; i++)
      #pragma unroll
      for (int j = 0; j < TJ; j++) {
        float* p = out + (size_t)(m0 + wr * 32 + i * 16 + 8 * half) * N + (ncol + j * 16);
        #pragma unroll
        for (int e = 0; e < 8; e++) {
          float vv = acc[i][j][e];
          if (RELU) vv = fmaxf(vv, 0.0f);
          *p = vv; p += N;
        }
      }
  } else {
    #pragma unroll
    for (int i = 0; i < TI; i++)
      #pragma unroll
      for (int j = 0; j < TJ; j++) {
        int mr0 = m0 + wr * 32 + i * 16 + 8 * half;
        float* p = out + (size_t)mr0 * N + (ncol + j * 16);
        #pragma unroll
        for (int e = 0; e < 8; e++) {
          if (mr0 + e < M) {
            float vv = acc[i][j][e];
            if (RELU) vv = fmaxf(vv, 0.0f);
            *p = vv;
          }
          p += N;
        }
      }
  }
}

// ---------------------------------------------------------------------------
// Self-attention: one wave per (b,h,query). scores->softmax->AV, HD=32 lanes.
// ---------------------------------------------------------------------------
__global__ __launch_bounds__(256) void attn_kernel(
    const float* __restrict__ q, const float* __restrict__ k,
    const float* __restrict__ v, float* __restrict__ sa) {
  __shared__ float sq[8][32];
  __shared__ float ps[8][LQv];
  const int wave = threadIdx.x >> 5, lane = threadIdx.x & 31;
  const int bh = blockIdx.x;
  const int b = bh >> 3, h = bh & 7;
  int qi = blockIdx.y * 8 + wave; if (qi > LQv - 1) qi = LQv - 1;  // uniform barriers

  const size_t qoff = ((size_t)(b * LQv + qi)) * Cv + h * HDv;
  const float scale = 0.17677669529663687f;   // 1/sqrt(32)
  sq[wave][lane] = q[qoff + lane] * scale;
  __syncthreads();

  float ml = -1e30f;
  for (int kk = lane; kk < LQv; kk += 32) {
    const float* kp = k + ((size_t)(b * LQv + kk)) * Cv + h * HDv;
    float dot = 0.f;
    #pragma unroll
    for (int d = 0; d < 32; d++) dot += sq[wave][d] * kp[d];
    ps[wave][kk] = dot;
    ml = fmaxf(ml, dot);
  }
  #pragma unroll
  for (int o = 16; o > 0; o >>= 1) ml = fmaxf(ml, __shfl_xor(ml, o));
  float sl = 0.f;
  for (int kk = lane; kk < LQv; kk += 32) {
    float e = __expf(ps[wave][kk] - ml);
    ps[wave][kk] = e;
    sl += e;
  }
  #pragma unroll
  for (int o = 16; o > 0; o >>= 1) sl += __shfl_xor(sl, o);
  const float inv = 1.0f / sl;
  __syncthreads();

  float acc = 0.f;
  #pragma unroll 4
  for (int kk = 0; kk < LQv; kk++)
    acc += ps[wave][kk] * v[((size_t)(b * LQv + kk)) * Cv + h * HDv + lane];
  sa[qoff + lane] = acc * inv;
}

// ---------------------------------------------------------------------------
// Deformable bilinear sampling. One wave per (b,q,h); lane = head dim.
// ---------------------------------------------------------------------------
__device__ __forceinline__ float fetch_val(const float* __restrict__ value,
                                           int b, int h, int lane, int st,
                                           int Wd, int H, int y, int x) {
  bool valid = (y >= 0) & (y < H) & (x >= 0) & (x < Wd);
  int yc = min(max(y, 0), H - 1), xc = min(max(x, 0), Wd - 1);
  size_t s = (size_t)b * Sv + st + yc * Wd + xc;
  float vv = value[s * Cv + h * HDv + lane];
  return valid ? vv : 0.f;
}

__global__ __launch_bounds__(256) void sample_kernel(
    const float* __restrict__ value, const float* __restrict__ offs,
    const float* __restrict__ awl, const float* __restrict__ refp,
    float* __restrict__ t2) {
  const int wave = threadIdx.x >> 5, lane = threadIdx.x & 31;
  const int gid = blockIdx.x * 8 + wave;        // exactly B2*LQ*NH = 24000
  const int b = gid / (LQv * NHv);
  const int rem = gid % (LQv * NHv);
  const int qi = rem >> 3, h = rem & 7;
  const size_t row = (size_t)b * LQv + qi;

  // softmax over the 16 (level,point) logits
  const float* wl = awl + row * 128 + h * 16;
  float lg[16]; float m = -1e30f;
  #pragma unroll
  for (int j = 0; j < 16; j++) { lg[j] = wl[j]; m = fmaxf(m, lg[j]); }
  float sum = 0.f;
  #pragma unroll
  for (int j = 0; j < 16; j++) { lg[j] = __expf(lg[j] - m); sum += lg[j]; }
  const float invs = 1.f / sum;

  const int LH[4] = {92, 46, 23, 12};
  const int LW[4] = {92, 46, 23, 12};
  const int LS[4] = {0, 8464, 10580, 11109};
  const float* op = offs + row * Cv + h * 32;   // (l,p,xy): ((l*4+p)*2+xy)
  const float* rp = refp + row * 8;             // (NL,2)

  float acc = 0.f;
  #pragma unroll
  for (int l = 0; l < 4; l++) {
    const float rx = rp[l * 2 + 0], ry = rp[l * 2 + 1];
    const int H = LH[l], Wd = LW[l], st = LS[l];
    #pragma unroll
    for (int p = 0; p < 4; p++) {
      float ox = op[(l * 4 + p) * 2 + 0], oy = op[(l * 4 + p) * 2 + 1];
      float x = (rx + ox / (float)Wd) * (float)Wd - 0.5f;
      float y = (ry + oy / (float)H) * (float)H - 0.5f;
      float x0f = floorf(x), y0f = floorf(y);
      float wx = x - x0f, wy = y - y0f;
      int x0 = (int)x0f, y0 = (int)y0f;
      float v00 = fetch_val(value, b, h, lane, st, Wd, H, y0,     x0);
      float v01 = fetch_val(value, b, h, lane, st, Wd, H, y0,     x0 + 1);
      float v10 = fetch_val(value, b, h, lane, st, Wd, H, y0 + 1, x0);
      float v11 = fetch_val(value, b, h, lane, st, Wd, H, y0 + 1, x0 + 1);
      float bil = v00 * (1 - wx) * (1 - wy) + v01 * wx * (1 - wy)
                + v10 * (1 - wx) * wy       + v11 * wx * wy;
      acc += lg[l * 4 + p] * invs * bil;
    }
  }
  t2[row * Cv + h * HDv + lane] = acc;
}

// ---------------------------------------------------------------------------
// out = LayerNorm(a + r) * g + b      (one 256-thread block per row, C=256)
// ---------------------------------------------------------------------------
__global__ __launch_bounds__(256) void add_ln_kernel(
    const float* __restrict__ a, const float* __restrict__ r,
    const float* __restrict__ g, const float* __restrict__ be,
    float* __restrict__ out) {
  __shared__ float red[8];
  const int tid = threadIdx.x, wave = tid >> 5, lane = tid & 31;
  const size_t off = (size_t)blockIdx.x * Cv + tid;
  float x = a[off] + r[off];

  float s = x;
  #pragma unroll
  for (int o = 16; o > 0; o >>= 1) s += __shfl_xor(s, o);
  if (lane == 0) red[wave] = s;
  __syncthreads();
  float tot = 0.f;
  #pragma unroll
  for (int i = 0; i < 8; i++) tot += red[i];
  const float mean = tot * (1.f / 256.f);
  __syncthreads();

  float d = x - mean;
  float s2 = d * d;
  #pragma unroll
  for (int o = 16; o > 0; o >>= 1) s2 += __shfl_xor(s2, o);
  if (lane == 0) red[wave] = s2;
  __syncthreads();
  float tot2 = 0.f;
  #pragma unroll
  for (int i = 0; i < 8; i++) tot2 += red[i];
  const float rstd = rsqrtf(tot2 * (1.f / 256.f) + 1e-5f);
  out[off] = d * rstd * g[tid] + be[tid];
}

__global__ __launch_bounds__(256) void add_kernel(
    const float* __restrict__ a, const float* __restrict__ b,
    float* __restrict__ o, int n) {
  int i = blockIdx.x * 256 + threadIdx.x;
  if (i < n) o[i] = a[i] + b[i];
}

// ---------------------------------------------------------------------------
extern "C" void kernel_launch(void* const* d_in, const int* in_sizes, int n_in,
                              void* d_out, int out_size, void* d_ws, size_t ws_size,
                              hipStream_t stream) {
  const float* tgt    = (const float*)d_in[0];
  const float* qpos   = (const float*)d_in[1];
  const float* refp   = (const float*)d_in[2];
  const float* src    = (const float*)d_in[3];
  // d_in[4] spatial shapes, d_in[5] level starts: compile-time constants here
  const float* in_proj_w  = (const float*)d_in[6];
  const float* in_proj_b  = (const float*)d_in[7];
  const float* out_proj_w = (const float*)d_in[8];
  const float* out_proj_b = (const float*)d_in[9];
  const float* norm1_g = (const float*)d_in[10];
  const float* norm1_b = (const float*)d_in[11];
  const float* norm2_g = (const float*)d_in[12];
  const float* norm2_b = (const float*)d_in[13];
  const float* norm3_g = (const float*)d_in[14];
  const float* norm3_b = (const float*)d_in[15];
  const float* samp_w  = (const float*)d_in[16];
  const float* samp_b  = (const float*)d_in[17];
  const float* attnw_w = (const float*)d_in[18];
  const float* attnw_b = (const float*)d_in[19];
  const float* value_w = (const float*)d_in[20];
  const float* value_b = (const float*)d_in[21];
  const float* outp_w  = (const float*)d_in[22];
  const float* outp_b  = (const float*)d_in[23];
  const float* lin1_w  = (const float*)d_in[24];
  const float* lin1_b  = (const float*)d_in[25];
  const float* lin2_w  = (const float*)d_in[26];
  const float* lin2_b  = (const float*)d_in[27];

  const size_t NE = (size_t)Mq * Cv;   // 768000
  float* ws = (float*)d_ws;
  size_t off = 0;
  auto nxt = [&](size_t n) { float* p = ws + off; off += n; return p; };
  float* qk  = nxt(NE);                 // later: query
  float* qb  = nxt(NE);                 // q,     later: sampling offsets
  float* kb  = nxt(NE);                 // k,     later: attn-weight logits
  float* vb  = nxt(NE);                 // v,     later: t2 (sampled)
  float* sab = nxt(NE);                 // sa,    later: t2 projected
  float* sa2 = nxt(NE);                 // sa proj, later: ffn out
  float* t1  = nxt(NE);
  float* t3  = nxt(NE);
  float* hb  = nxt((size_t)Mq * DFFv);  // FFN hidden
  float* val = nxt((size_t)Mval * Cv);  // projected value (115 MB)

  // BN=256 config: whole N in one block column -> A streams once from HBM
  auto gemm256 = [&](const float* A, const float* W, const float* Bias, float* O,
                     int M, int N, int K, bool relu) {
    dim3 grid(N / 256, (M + 63) / 64);
    if (relu) gemm_bf16_kernel<64, 256, 1><<<grid, 256, 0, stream>>>(A, W, Bias, O, M, N, K);
    else      gemm_bf16_kernel<64, 256, 0><<<grid, 256, 0, stream>>>(A, W, Bias, O, M, N, K);
  };
  // BN=64 config for N==128
  auto gemm64 = [&](const float* A, const float* W, const float* Bias, float* O,
                    int M, int N, int K) {
    dim3 grid(N / 64, (M + 127) / 128);
    gemm_bf16_kernel<128, 64, 0><<<grid, 256, 0, stream>>>(A, W, Bias, O, M, N, K);
  };

  // 1) qk = tgt + query_pos ; q/k/v projections
  add_kernel<<<(int)(NE / 256), 256, 0, stream>>>(tgt, qpos, qk, (int)NE);
  gemm256(qk,  in_proj_w,               in_proj_b,          qb, Mq, Cv, Cv, false);
  gemm256(qk,  in_proj_w + Cv * Cv,     in_proj_b + Cv,     kb, Mq, Cv, Cv, false);
  gemm256(tgt, in_proj_w + 2 * Cv * Cv, in_proj_b + 2 * Cv, vb, Mq, Cv, Cv, false);

  // 2) self-attention + out_proj + LN(norm2)
  attn_kernel<<<dim3(B2v * NHv, (LQv + 7) / 8), 256, 0, stream>>>(qb, kb, vb, sab);
  gemm256(sab, out_proj_w, out_proj_b, sa2, Mq, Cv, Cv, false);
  add_ln_kernel<<<Mq, 256, 0, stream>>>(tgt, sa2, norm2_g, norm2_b, t1);

  // 3) deformable attention: offsets, weights, value projection, sampling
  add_kernel<<<(int)(NE / 256), 256, 0, stream>>>(t1, qpos, qk, (int)NE); // query
  gemm256(qk,  samp_w,  samp_b,  qb, Mq, Cv,  Cv, false);   // sampling offsets
  gemm64 (qk,  attnw_w, attnw_b, kb, Mq, 128, Cv);          // attn-weight logits
  gemm256(src, value_w, value_b, val, Mval, Cv, Cv, false); // big GEMM, A read once
  sample_kernel<<<Mq, 256, 0, stream>>>(val, qb, kb, refp, vb);           // t2
  gemm256(vb, outp_w, outp_b, sab, Mq, Cv, Cv, false);
  add_ln_kernel<<<Mq, 256, 0, stream>>>(t1, sab, norm1_g, norm1_b, t3);

  // 4) FFN + LN(norm3) -> d_out
  gemm256(t3, lin1_w, lin1_b, hb, Mq, DFFv, Cv, true);
  gemm256(hb, lin2_w, lin2_b, sa2, Mq, Cv, DFFv, false);
  add_ln_kernel<<<Mq, 256, 0, stream>>>(t3, sa2, norm3_g, norm3_b, (float*)d_out);
}
